// FormulaNetEdges_88493506167072
// MI455X (gfx1250) — matrix operand order repaired and checked by reading, not compiled, not run on hardware
//
#include <hip/hip_runtime.h>

#define DEV __device__ __forceinline__

typedef __attribute__((ext_vector_type(16))) __bf16 v16bf;
typedef __attribute__((ext_vector_type(8)))  float  v8f;
typedef __attribute__((ext_vector_type(4)))  unsigned int v4u;
typedef __attribute__((ext_vector_type(4)))  unsigned int u32x4;
typedef __attribute__((ext_vector_type(8)))  int i32x8;
typedef __attribute__((ext_vector_type(4)))  int i32x4;

union Frag16 { v4u u[2]; v16bf v; };

DEV unsigned short f2bf(float f) {
  unsigned u = __float_as_uint(f);
  u += 0x7FFFu + ((u >> 16) & 1u);          // round-to-nearest-even-ish
  return (unsigned short)(u >> 16);
}
DEV float bf2f(unsigned short h) { return __uint_as_float(((unsigned)h) << 16); }

// 16-element bf16 fragment: elements 0-7 at p, elements 8-15 at p+gap
DEV v16bf load_frag(const unsigned short* p, int gap) {
  Frag16 f;
  f.u[0] = *(const v4u*)(p);
  f.u[1] = *(const v4u*)(p + gap);
  return f.v;
}

// ---- Tensor Data Mover: stage a [rows, k_elems] bf16 panel (contiguous rows) into LDS ----
// D# per CDNA5 ISA: group0 = {flags, lds_addr, global_addr, type=2},
// group1 = {data_size=2B, tensor_dim0=k, tensor_dim1=rows, tile_dim0=k, tile_dim1=rows, stride0=k}
DEV void tdm_load_panel(void* lds_dst, const void* gsrc, int k_elems, int rows) {
  unsigned ldsa = (unsigned)(unsigned long long)lds_dst;   // generic LDS addr: low 32 bits = LDS offset
  unsigned long long ga = (unsigned long long)gsrc;
  u32x4 g0;
  g0.x = 1u;                                              // count=1, user descriptor
  g0.y = ldsa;                                            // lds_addr
  g0.z = (unsigned)ga;                                    // global_addr[31:0]
  g0.w = (unsigned)((ga >> 32) & 0x01FFFFFFull) | (2u << 30);  // global_addr[56:32] | type=2
  i32x8 g1;
  g1[0] = (int)(1u << 16);                                // data_size=1 -> 2 bytes; no multicast
  g1[1] = (int)(((unsigned)k_elems & 0xFFFFu) << 16);     // tensor_dim0[15:0] (bits 63:48)
  g1[2] = (int)(((unsigned)rows & 0xFFFFu) << 16);        // tensor_dim1[15:0] (bits 95:80)
  g1[3] = (int)(((unsigned)k_elems & 0xFFFFu) << 16);     // tile_dim0 (bits 127:112)
  g1[4] = (int)((unsigned)rows & 0xFFFFu);                // tile_dim1 (bits 143:128), tile_dim2=0
  g1[5] = (int)(unsigned)k_elems;                         // tensor_dim0_stride[31:0]
  g1[6] = 0;
  g1[7] = 0;
  i32x4 z4 = {};
#if defined(__clang_major__) && (__clang_major__ >= 23)
  i32x8 z8 = {};
  __builtin_amdgcn_tensor_load_to_lds(g0, g1, z4, z4, z8, 0);
#else
  __builtin_amdgcn_tensor_load_to_lds(g0, g1, z4, z4, 0);
#endif
}

// ---------------- elementwise helpers ----------------
__global__ void k_zero_f32(float* p, long n) {
  long i = (long)blockIdx.x * 256 + threadIdx.x;
  if (i < n) p[i] = 0.0f;
}
__global__ void k_cvt_bf16(const float* __restrict__ s, unsigned short* __restrict__ d, long n) {
  long i = (long)blockIdx.x * 256 + threadIdx.x;
  if (i < n) d[i] = f2bf(s[i]);
}
// src f32 [K,Nout] row-major -> dst bf16 [Nout,K] (transposed so B loads are contiguous)
__global__ void k_tcvt_bf16(const float* __restrict__ s, unsigned short* __restrict__ d, int K, int Nn) {
  long i = (long)blockIdx.x * 256 + threadIdx.x;
  if (i < (long)K * Nn) {
    int k = (int)(i / Nn), n = (int)(i % Nn);
    d[(long)n * K + k] = f2bf(s[i]);
  }
}
__global__ void k_deg_count(const int* __restrict__ src, float* __restrict__ deg, int E) {
  int i = blockIdx.x * 256 + threadIdx.x;
  if (i < E) atomicAdd(&deg[src[i]], 1.0f);
}
__global__ void k_deg_inv(float* deg, int n) {
  int i = blockIdx.x * 256 + threadIdx.x;
  if (i < n) { float d = deg[i]; deg[i] = d > 0.0f ? 1.0f / d : 0.0f; }
}
__global__ void k_scatter_add(const unsigned short* __restrict__ msg, const int* __restrict__ tgt,
                              float* __restrict__ acc, long n /* = E*128 */) {
  long i = (long)blockIdx.x * 256 + threadIdx.x;
  if (i < n) {
    int e = (int)(i >> 7), c = (int)(i & 127);
    atomicAdd(&acc[(long)tgt[e] * 128 + c], bf2f(msg[i]));
  }
}
__global__ void k_scale_cvt(const float* __restrict__ acc, const float* __restrict__ dinv,
                            unsigned short* __restrict__ out, long n) {
  long i = (long)blockIdx.x * 256 + threadIdx.x;
  if (i < n) out[i] = f2bf(acc[i] * dinv[i >> 7]);
}
__global__ void k_res_add(unsigned short* __restrict__ nodes, const unsigned short* __restrict__ upd, long n) {
  long i = (long)blockIdx.x * 256 + threadIdx.x;
  if (i < n) nodes[i] = f2bf(bf2f(nodes[i]) + bf2f(upd[i]));
}

// ---------------- WMMA GEMM kernels ----------------
// Common scheme: 8 waves/block, each wave owns one 16-row tile and holds its full
// 16xK A tile in registers (K/32 fragments). B is staged per 64-column chunk into
// LDS by the TDM (wave 0 issues tensor_load_to_lds, waits TENSORcnt, block barrier),
// then read back as ds_load_b128 fragments. Stores guarded for tail tiles; EXEC
// stays all-ones for every WMMA.

// out[M,Nout] = relu(A[idx? ,K] @ Wt^T + bias)
template<int K, int CH>
__global__ __launch_bounds__(256) void k_gemm_relu_t(
    const unsigned short* __restrict__ A, const int* __restrict__ idx,
    const unsigned short* __restrict__ Wt, const float* __restrict__ bias,
    unsigned short* __restrict__ out, int M, int Nout)
{
  constexpr int NF = K / 32;
  __shared__ __align__(16) unsigned short sW[CH * K];
  const int lane = threadIdx.x & 31;
  const int wv   = threadIdx.x >> 5;
  const int row0 = (blockIdx.x * 8 + wv) * 16;
  const bool valid = row0 < M;
  const int hf  = lane >> 4;
  const int l15 = lane & 15;
  const int r   = valid ? row0 + l15 : 0;
  const long arow = (long)(idx ? idx[r] : r) * K;

  v16bf a[NF];
#pragma unroll
  for (int f = 0; f < NF; f++)
    a[f] = load_frag(A + arow + f * 32 + hf * 8, 16);

  for (int c0 = 0; c0 < Nout; c0 += CH) {
    __syncthreads();                       // previous chunk fully consumed
    if (wv == 0) {
      tdm_load_panel(sW, Wt + (size_t)c0 * K, K, CH);
      __builtin_amdgcn_s_wait_tensorcnt(0);
    }
    __syncthreads();                       // chunk visible to all waves
#pragma unroll
    for (int ct = 0; ct < CH / 16; ct++) {
      const unsigned short* wrow = sW + (size_t)(ct * 16 + l15) * K;
      v8f acc = {};
#pragma unroll
      for (int f = 0; f < NF; f++) {
        v16bf b = load_frag(wrow + f * 32 + hf * 8, 16);
        acc = __builtin_amdgcn_wmma_f32_16x16x32_bf16(false, a[f], false, b, (short)0, acc, false, false);
      }
      if (valid) {
        const int c = c0 + ct * 16 + l15;
        const float bv = bias[c];
#pragma unroll
        for (int i = 0; i < 8; i++) {
          float v = acc[i] + bv;
          v = v > 0.0f ? v : 0.0f;
          out[(long)(row0 + i + hf * 8) * Nout + c] = f2bf(v);
        }
      }
    }
  }
}

// A row r = concat(x0[idx0?r], x1[idx1?r], x2[r]); segments 128 wide, K=384.
template<int CH>
__global__ __launch_bounds__(256) void k_gemm_cat3_t(
    const unsigned short* __restrict__ x0, const int* __restrict__ idx0,
    const unsigned short* __restrict__ x1, const int* __restrict__ idx1,
    const unsigned short* __restrict__ x2,
    const unsigned short* __restrict__ Wt, const float* __restrict__ bias,
    unsigned short* __restrict__ out, int M, int Nout)
{
  constexpr int K = 384, NF = 12;
  __shared__ __align__(16) unsigned short sW[CH * K];
  const int lane = threadIdx.x & 31;
  const int wv   = threadIdx.x >> 5;
  const int row0 = (blockIdx.x * 8 + wv) * 16;
  const bool valid = row0 < M;
  const int hf  = lane >> 4;
  const int l15 = lane & 15;
  const int r   = valid ? row0 + l15 : 0;
  const long r0 = (long)(idx0 ? idx0[r] : r) * 128;
  const long r1 = (long)(idx1 ? idx1[r] : r) * 128;
  const long r2 = (long)r * 128;

  v16bf a[NF];
#pragma unroll
  for (int f = 0; f < NF; f++) {
    const int seg = f >> 2;                // 4 fragments per 128-wide segment
    const unsigned short* xp = (seg == 0) ? x0 : (seg == 1) ? x1 : x2;
    const long rb = (seg == 0) ? r0 : (seg == 1) ? r1 : r2;
    a[f] = load_frag(xp + rb + (f & 3) * 32 + hf * 8, 16);
  }

  for (int c0 = 0; c0 < Nout; c0 += CH) {
    __syncthreads();
    if (wv == 0) {
      tdm_load_panel(sW, Wt + (size_t)c0 * K, K, CH);
      __builtin_amdgcn_s_wait_tensorcnt(0);
    }
    __syncthreads();
#pragma unroll
    for (int ct = 0; ct < CH / 16; ct++) {
      const unsigned short* wrow = sW + (size_t)(ct * 16 + l15) * K;
      v8f acc = {};
#pragma unroll
      for (int f = 0; f < NF; f++) {
        v16bf b = load_frag(wrow + f * 32 + hf * 8, 16);
        acc = __builtin_amdgcn_wmma_f32_16x16x32_bf16(false, a[f], false, b, (short)0, acc, false, false);
      }
      if (valid) {
        const int c = c0 + ct * 16 + l15;
        const float bv = bias[c];
#pragma unroll
        for (int i = 0; i < 8; i++) {
          float v = acc[i] + bv;
          v = v > 0.0f ? v : 0.0f;
          out[(long)(row0 + i + hf * 8) * Nout + c] = f2bf(v);
        }
      }
    }
  }
}

// Final layer fused with segment_max (post-ReLU >= 0; out zero-inited; int atomicMax == float max)
template<int K, int CH>
__global__ __launch_bounds__(256) void k_gemm_segmax_t(
    const unsigned short* __restrict__ A, const unsigned short* __restrict__ Wt,
    const float* __restrict__ bias, const int* __restrict__ batch,
    float* __restrict__ out, int M, int Nout)
{
  constexpr int NF = K / 32;
  __shared__ __align__(16) unsigned short sW[CH * K];
  const int lane = threadIdx.x & 31;
  const int wv   = threadIdx.x >> 5;
  const int row0 = (blockIdx.x * 8 + wv) * 16;
  const bool valid = row0 < M;
  const int hf  = lane >> 4;
  const int l15 = lane & 15;
  const long arow = (long)(valid ? row0 + l15 : 0) * K;

  v16bf a[NF];
#pragma unroll
  for (int f = 0; f < NF; f++)
    a[f] = load_frag(A + arow + f * 32 + hf * 8, 16);

  for (int c0 = 0; c0 < Nout; c0 += CH) {
    __syncthreads();
    if (wv == 0) {
      tdm_load_panel(sW, Wt + (size_t)c0 * K, K, CH);
      __builtin_amdgcn_s_wait_tensorcnt(0);
    }
    __syncthreads();
#pragma unroll
    for (int ct = 0; ct < CH / 16; ct++) {
      const unsigned short* wrow = sW + (size_t)(ct * 16 + l15) * K;
      v8f acc = {};
#pragma unroll
      for (int f = 0; f < NF; f++) {
        v16bf b = load_frag(wrow + f * 32 + hf * 8, 16);
        acc = __builtin_amdgcn_wmma_f32_16x16x32_bf16(false, a[f], false, b, (short)0, acc, false, false);
      }
      if (valid) {
        const int c = c0 + ct * 16 + l15;
        const float bv = bias[c];
#pragma unroll
        for (int i = 0; i < 8; i++) {
          float v = acc[i] + bv;
          v = v > 0.0f ? v : 0.0f;
          const int g = batch[row0 + i + hf * 8];
          atomicMax((int*)&out[(long)g * Nout + c], __float_as_int(v));
        }
      }
    }
  }
}

// ---------------- host ----------------
extern "C" void kernel_launch(void* const* d_in, const int* in_sizes, int n_in,
                              void* d_out, int out_size, void* d_ws, size_t ws_size,
                              hipStream_t stream) {
  (void)in_sizes; (void)n_in; (void)out_size; (void)ws_size;
  const int N = 50000, E = 200000, V = 2000, ME = 3, G = 256, ITERS = 2;

  const int* x     = (const int*)d_in[0];
  const int* src   = (const int*)d_in[1];          // edge_index[0]
  const int* dst   = src + E;                       // edge_index[1]
  const int* eattr = (const int*)d_in[2];
  const int* batch = (const int*)d_in[3];
  const float* emb_node = (const float*)d_in[4];
  const float* Wn1 = (const float*)d_in[5];  const float* bn1 = (const float*)d_in[6];
  const float* Wn2 = (const float*)d_in[7];  const float* bn2 = (const float*)d_in[8];
  const float* emb_edge = (const float*)d_in[9];
  const float* We1 = (const float*)d_in[10]; const float* be1 = (const float*)d_in[11];
  const float* We2 = (const float*)d_in[12]; const float* be2 = (const float*)d_in[13];
  const float *W1[3], *B1[3], *W2[3], *B2[3], *W3[3], *B3[3];
  for (int t = 0; t < 3; t++) {
    int base = 14 + 6 * t;
    W1[t] = (const float*)d_in[base + 0]; B1[t] = (const float*)d_in[base + 1];
    W2[t] = (const float*)d_in[base + 2]; B2[t] = (const float*)d_in[base + 3];
    W3[t] = (const float*)d_in[base + 4]; B3[t] = (const float*)d_in[base + 5];
  }
  const float* Wo1 = (const float*)d_in[32]; const float* bo1 = (const float*)d_in[33];
  const float* Wo2 = (const float*)d_in[34]; const float* bo2 = (const float*)d_in[35];

  // workspace carve-up (256B aligned slices)
  size_t off = 0;
  char* wsb = (char*)d_ws;
  auto alloc = [&](size_t bytes) -> void* {
    void* p = wsb + off; off += (bytes + 255) & ~(size_t)255; return p;
  };
  unsigned short* embNb = (unsigned short*)alloc((size_t)V * 256 * 2);
  unsigned short* embEb = (unsigned short*)alloc((size_t)ME * 256 * 2);
  unsigned short* Wn1t = (unsigned short*)alloc((size_t)256 * 128 * 2);
  unsigned short* Wn2t = (unsigned short*)alloc((size_t)128 * 128 * 2);
  unsigned short* We1t = (unsigned short*)alloc((size_t)256 * 128 * 2);
  unsigned short* We2t = (unsigned short*)alloc((size_t)128 * 128 * 2);
  unsigned short *W1t[3], *W2t[3], *W3t[3];
  for (int t = 0; t < 3; t++) {
    W1t[t] = (unsigned short*)alloc((size_t)384 * 256 * 2);
    W2t[t] = (unsigned short*)alloc((size_t)256 * 128 * 2);
    W3t[t] = (unsigned short*)alloc((size_t)128 * 128 * 2);
  }
  unsigned short* Wo1t = (unsigned short*)alloc((size_t)128 * 512 * 2);
  unsigned short* Wo2t = (unsigned short*)alloc((size_t)512 * 1024 * 2);
  unsigned short* nodes = (unsigned short*)alloc((size_t)N * 128 * 2);
  unsigned short* ea    = (unsigned short*)alloc((size_t)E * 128 * 2);
  unsigned short* bufA  = (unsigned short*)alloc((size_t)E * 256 * 2);  // also h1[N,512]
  unsigned short* bufB  = (unsigned short*)alloc((size_t)E * 128 * 2);
  unsigned short* bufC  = (unsigned short*)alloc((size_t)E * 128 * 2);  // msg / upd
  float* fi   = (float*)alloc((size_t)N * 128 * 4);
  float* fo   = (float*)alloc((size_t)N * 128 * 4);
  unsigned short* fib = (unsigned short*)alloc((size_t)N * 128 * 2);
  unsigned short* fob = (unsigned short*)alloc((size_t)N * 128 * 2);
  float* deg  = (float*)alloc((size_t)N * 4);

  const dim3 TPB(256);
  auto cdiv = [](long a, long b) { return (unsigned)((a + b - 1) / b); };
  auto ggrid = [&](int M) { return dim3(cdiv((M + 15) / 16, 8)); };  // 8 row tiles / block

  // --- weight conversion (deterministic, every call) ---
  k_cvt_bf16<<<cdiv((long)V * 256, 256), TPB, 0, stream>>>(emb_node, embNb, (long)V * 256);
  k_cvt_bf16<<<cdiv((long)ME * 256, 256), TPB, 0, stream>>>(emb_edge, embEb, (long)ME * 256);
  auto tc = [&](const float* s, unsigned short* d, int K, int Nn) {
    k_tcvt_bf16<<<cdiv((long)K * Nn, 256), TPB, 0, stream>>>(s, d, K, Nn);
  };
  tc(Wn1, Wn1t, 256, 128); tc(Wn2, Wn2t, 128, 128);
  tc(We1, We1t, 256, 128); tc(We2, We2t, 128, 128);
  for (int t = 0; t < 3; t++) {
    tc(W1[t], W1t[t], 384, 256); tc(W2[t], W2t[t], 256, 128); tc(W3[t], W3t[t], 128, 128);
  }
  tc(Wo1, Wo1t, 128, 512); tc(Wo2, Wo2t, 512, 1024);

  // --- encoders ---
  k_gemm_relu_t<256, 64><<<ggrid(N), TPB, 0, stream>>>(embNb, x, Wn1t, bn1, bufB, N, 128);
  k_gemm_relu_t<128, 64><<<ggrid(N), TPB, 0, stream>>>(bufB, nullptr, Wn2t, bn2, nodes, N, 128);
  k_gemm_relu_t<256, 64><<<ggrid(E), TPB, 0, stream>>>(embEb, eattr, We1t, be1, bufB, E, 128);
  k_gemm_relu_t<128, 64><<<ggrid(E), TPB, 0, stream>>>(bufB, nullptr, We2t, be2, ea, E, 128);

  // --- degree (over src), then in-place inverse ---
  k_zero_f32<<<cdiv(N, 256), TPB, 0, stream>>>(deg, N);
  k_deg_count<<<cdiv(E, 256), TPB, 0, stream>>>(src, deg, E);
  k_deg_inv<<<cdiv(N, 256), TPB, 0, stream>>>(deg, N);

  // --- message-passing iterations ---
  for (int it = 0; it < ITERS; it++) {
    // parent: concat(nodes[dst], nodes[src], ea) -> sum at dst
    k_gemm_cat3_t<64><<<ggrid(E), TPB, 0, stream>>>(nodes, dst, nodes, src, ea, W1t[0], B1[0], bufA, E, 256);
    k_gemm_relu_t<256, 64><<<ggrid(E), TPB, 0, stream>>>(bufA, nullptr, W2t[0], B2[0], bufB, E, 128);
    k_gemm_relu_t<128, 64><<<ggrid(E), TPB, 0, stream>>>(bufB, nullptr, W3t[0], B3[0], bufC, E, 128);
    k_zero_f32<<<cdiv((long)N * 128, 256), TPB, 0, stream>>>(fi, (long)N * 128);
    k_scatter_add<<<cdiv((long)E * 128, 256), TPB, 0, stream>>>(bufC, dst, fi, (long)E * 128);
    // child: concat(nodes[src], nodes[dst], ea) -> sum at src
    k_gemm_cat3_t<64><<<ggrid(E), TPB, 0, stream>>>(nodes, src, nodes, dst, ea, W1t[1], B1[1], bufA, E, 256);
    k_gemm_relu_t<256, 64><<<ggrid(E), TPB, 0, stream>>>(bufA, nullptr, W2t[1], B2[1], bufB, E, 128);
    k_gemm_relu_t<128, 64><<<ggrid(E), TPB, 0, stream>>>(bufB, nullptr, W3t[1], B3[1], bufC, E, 128);
    k_zero_f32<<<cdiv((long)N * 128, 256), TPB, 0, stream>>>(fo, (long)N * 128);
    k_scatter_add<<<cdiv((long)E * 128, 256), TPB, 0, stream>>>(bufC, src, fo, (long)E * 128);
    // normalize by deg_inv (both branches use src-degree, as in reference)
    k_scale_cvt<<<cdiv((long)N * 128, 256), TPB, 0, stream>>>(fi, deg, fib, (long)N * 128);
    k_scale_cvt<<<cdiv((long)N * 128, 256), TPB, 0, stream>>>(fo, deg, fob, (long)N * 128);
    // node update MLP + residual
    k_gemm_cat3_t<64><<<ggrid(N), TPB, 0, stream>>>(nodes, nullptr, fib, nullptr, fob, W1t[2], B1[2], bufA, N, 256);
    k_gemm_relu_t<256, 64><<<ggrid(N), TPB, 0, stream>>>(bufA, nullptr, W2t[2], B2[2], bufB, N, 128);
    k_gemm_relu_t<128, 64><<<ggrid(N), TPB, 0, stream>>>(bufB, nullptr, W3t[2], B3[2], bufC, N, 128);
    k_res_add<<<cdiv((long)N * 128, 256), TPB, 0, stream>>>(nodes, bufC, (long)N * 128);
  }

  // --- output MLP + fused segment_max pool ---
  k_gemm_relu_t<128, 64><<<ggrid(N), TPB, 0, stream>>>(nodes, nullptr, Wo1t, bo1, bufA, N, 512);
  k_zero_f32<<<cdiv((long)G * 1024, 256), TPB, 0, stream>>>((float*)d_out, (long)G * 1024);
  k_gemm_segmax_t<512, 64><<<ggrid(N), TPB, 0, stream>>>(bufA, Wo2t, bo2, batch, (float*)d_out, N, 1024);
}
